// STAttention_notf_v5_11690900980416
// MI455X (gfx1250) — matrix operand order repaired and checked by reading, not compiled, not run on hardware
//
#include <hip/hip_runtime.h>
#include <hip/hip_fp16.h>

typedef __attribute__((ext_vector_type(16))) _Float16 v16h;
typedef __attribute__((ext_vector_type(8)))  float    v8f;

#define DEV static __device__ __forceinline__

constexpr int Hd = 128;   // hidden
constexpr int Fd = 64;    // feature
constexpr int BT = 128;   // batch tile per workgroup (8 waves x 16 rows-worth of tiles)

DEV v8f wmma16(v16h a, v16h b, v8f c) {
  // D = A(16x32 f16) x B(32x16 f16) + C(16x16 f32)
  return __builtin_amdgcn_wmma_f32_16x16x32_f16(false, a, false, b, (short)0, c, false, false);
}

// A fragment (16x32, f16, row-major source, leading dim ld).
// ISA layout: lane m (0..15) holds row m; lane-group lg selects K-halves;
// vgpr v -> K = (v>>2)*16 + lg*8 + (v&3)*2 (pairs).
DEV v16h frag_a_h(const _Float16* base, int ld) {
  const int lane = threadIdx.x & 31;
  const _Float16* p = base + (lane & 15) * ld;
  const int lg = lane >> 4;
  v16h a;
#pragma unroll
  for (int v = 0; v < 8; ++v) {
    const int k = ((v >> 2) << 4) + (lg << 3) + ((v & 3) << 1);
    a[2 * v] = p[k];
    a[2 * v + 1] = p[k + 1];
  }
  return a;
}

// Same A layout, converting from f32 source (used for f32 c-state / dec_in).
DEV v16h frag_a_f(const float* base, int ld) {
  const int lane = threadIdx.x & 31;
  const float* p = base + (lane & 15) * ld;
  const int lg = lane >> 4;
  v16h a;
#pragma unroll
  for (int v = 0; v < 8; ++v) {
    const int k = ((v >> 2) << 4) + (lg << 3) + ((v & 3) << 1);
    a[2 * v] = (_Float16)p[k];
    a[2 * v + 1] = (_Float16)p[k + 1];
  }
  return a;
}

// B = W^T fragment for out = X @ W^T. W row-major [N_total, K_total]; caller
// pre-offsets W to (n0, k0). Lane n = output column; lane-group lg picks K
// octets: K = lg*8 + 0..7 and 16 + lg*8 + 0..7 (two b128 loads).
DEV v16h frag_bT(const _Float16* W, int ld) {
  const int lane = threadIdx.x & 31;
  const _Float16* p = W + (lane & 15) * ld + ((lane >> 4) << 3);
  v16h b;
#pragma unroll
  for (int v = 0; v < 16; ++v) b[v] = p[v < 8 ? v : v + 8];
  return b;
}

DEV float sigm(float x) { return 1.0f / (1.0f + expf(-x)); }

// ---------------- weight prep ----------------
__global__ void cvt_f16_kern(const float* __restrict__ s, _Float16* __restrict__ d, int n) {
  int i = blockIdx.x * blockDim.x + threadIdx.x;
  if (i < n) d[i] = (_Float16)s[i];
}
__global__ void bias2_kern(const float* __restrict__ a, const float* __restrict__ b,
                           float* __restrict__ d, int n) {
  int i = blockIdx.x * blockDim.x + threadIdx.x;
  if (i < n) d[i] = a[i] + b[i];
}

// ---------------- encoder: spatial-attention LSTMCell, fused over Te ----------------
__global__ void __launch_bounds__(256)
enc_kernel(const float* __restrict__ xall,        // [B,Te,F]
           const _Float16* __restrict__ Wi16,     // [H,F]
           const float* __restrict__ Wib,         // [H]
           const _Float16* __restrict__ We16,     // [H,2H]
           const _Float16* __restrict__ Vd16,     // [F,H]
           const float* __restrict__ Vdb,         // [F]
           const _Float16* __restrict__ Wih16,    // [4H,F]
           const _Float16* __restrict__ Whh16,    // [4H,H]
           const float* __restrict__ eb,          // [4H] = bih+bhh
           _Float16* __restrict__ mid,            // [Te,B,H]
           int B, int Te) {
  extern __shared__ __align__(16) char smem[];
  _Float16* sh_h0 = (_Float16*)smem;                  // BT*H f16
  _Float16* sh_h1 = sh_h0 + BT * Hd;                  // BT*H f16 (ping-pong)
  float* sf_c = (float*)(sh_h1 + BT * Hd);            // BT*H f32
  _Float16* sh_x = (_Float16*)(sf_c + BT * Hd);       // BT*F f16
  _Float16* sh_t = sh_x + BT * Fd;                    // BT*H f16 tanh buffer
  float* sf_s = (float*)(sh_t + BT * Hd);             // BT*F f32 scores
  _Float16* sh_xin = (_Float16*)(sf_s + BT * Fd);     // BT*F f16 attended x

  const int tid = threadIdx.x;
  const int w = tid >> 5;
  const int lane = tid & 31;
  const int nn = lane & 15;
  const int lg = lane >> 4;
  const int b0 = blockIdx.x * BT;

  for (int i = tid; i < BT * Hd; i += 256) { sh_h0[i] = (_Float16)0.0f; sf_c[i] = 0.0f; }
  __syncthreads();

  _Float16* hcur = sh_h0;
  _Float16* hnxt = sh_h1;

  for (int t = 0; t < Te; ++t) {
    // stage x_t into LDS (f16)
    for (int i = tid; i < BT * Fd; i += 256) {
      int r = i >> 6, f = i & 63;
      sh_x[i] = (_Float16)xall[((size_t)(b0 + r) * Te + t) * Fd + f];
    }
    __syncthreads();

    // phase A: T = tanh([h,c] @ We^T + x @ Wi^T + Wib)   [BT,H]
    {
      const int n0 = w * 16;
      for (int rt = 0; rt < 8; ++rt) {
        v8f acc = {};
#pragma unroll
        for (int k0 = 0; k0 < Hd; k0 += 32)
          acc = wmma16(frag_a_h(hcur + rt * 16 * Hd + k0, Hd),
                       frag_bT(We16 + n0 * (2 * Hd) + k0, 2 * Hd), acc);
#pragma unroll
        for (int k0 = 0; k0 < Hd; k0 += 32)
          acc = wmma16(frag_a_f(sf_c + rt * 16 * Hd + k0, Hd),
                       frag_bT(We16 + n0 * (2 * Hd) + Hd + k0, 2 * Hd), acc);
#pragma unroll
        for (int k0 = 0; k0 < Fd; k0 += 32)
          acc = wmma16(frag_a_h(sh_x + rt * 16 * Fd + k0, Fd),
                       frag_bT(Wi16 + n0 * Fd + k0, Fd), acc);
        const float bias = Wib[n0 + nn];
#pragma unroll
        for (int r = 0; r < 8; ++r) {
          const int row = rt * 16 + r + 8 * lg;
          sh_t[row * Hd + n0 + nn] = (_Float16)tanhf(acc[r] + bias);
        }
      }
    }
    __syncthreads();

    // phase B: s = T @ Vd^T + Vdb   [BT,F] f32
    {
      const int n0 = (w & 3) * 16;
      const int rtb = (w >> 2) * 4;
      for (int rt = rtb; rt < rtb + 4; ++rt) {
        v8f acc = {};
#pragma unroll
        for (int k0 = 0; k0 < Hd; k0 += 32)
          acc = wmma16(frag_a_h(sh_t + rt * 16 * Hd + k0, Hd),
                       frag_bT(Vd16 + n0 * Hd + k0, Hd), acc);
        const float bias = Vdb[n0 + nn];
#pragma unroll
        for (int r = 0; r < 8; ++r) {
          const int row = rt * 16 + r + 8 * lg;
          sf_s[row * Fd + n0 + nn] = acc[r] + bias;
        }
      }
    }
    __syncthreads();

    // phase C: row softmax over F, x_in = x * p
    if (tid < BT) {
      const int row = tid;
      float mx = -1e30f;
      for (int j = 0; j < Fd; ++j) mx = fmaxf(mx, sf_s[row * Fd + j]);
      float sum = 0.0f;
      for (int j = 0; j < Fd; ++j) {
        float e = expf(sf_s[row * Fd + j] - mx);
        sf_s[row * Fd + j] = e;
        sum += e;
      }
      const float inv = 1.0f / sum;
      for (int j = 0; j < Fd; ++j)
        sh_xin[row * Fd + j] = (_Float16)((float)sh_x[row * Fd + j] * sf_s[row * Fd + j] * inv);
    }
    __syncthreads();

    // phase D: LSTM cell  G = x_in@Wih^T + h@Whh^T + b ; update c,h
    {
      const int u0 = w * 16;
      for (int rt = 0; rt < 8; ++rt) {
        v8f ai = {}, af = {}, ag = {}, ao = {};
#pragma unroll
        for (int k0 = 0; k0 < Fd; k0 += 32) {
          v16h a = frag_a_h(sh_xin + rt * 16 * Fd + k0, Fd);
          ai = wmma16(a, frag_bT(Wih16 + (0 * Hd + u0) * Fd + k0, Fd), ai);
          af = wmma16(a, frag_bT(Wih16 + (1 * Hd + u0) * Fd + k0, Fd), af);
          ag = wmma16(a, frag_bT(Wih16 + (2 * Hd + u0) * Fd + k0, Fd), ag);
          ao = wmma16(a, frag_bT(Wih16 + (3 * Hd + u0) * Fd + k0, Fd), ao);
        }
#pragma unroll
        for (int k0 = 0; k0 < Hd; k0 += 32) {
          v16h a = frag_a_h(hcur + rt * 16 * Hd + k0, Hd);
          ai = wmma16(a, frag_bT(Whh16 + (0 * Hd + u0) * Hd + k0, Hd), ai);
          af = wmma16(a, frag_bT(Whh16 + (1 * Hd + u0) * Hd + k0, Hd), af);
          ag = wmma16(a, frag_bT(Whh16 + (2 * Hd + u0) * Hd + k0, Hd), ag);
          ao = wmma16(a, frag_bT(Whh16 + (3 * Hd + u0) * Hd + k0, Hd), ao);
        }
        const float bi = eb[0 * Hd + u0 + nn];
        const float bf = eb[1 * Hd + u0 + nn];
        const float bg = eb[2 * Hd + u0 + nn];
        const float bo = eb[3 * Hd + u0 + nn];
#pragma unroll
        for (int r = 0; r < 8; ++r) {
          const int row = rt * 16 + r + 8 * lg;
          const int col = u0 + nn;
          const float cold = sf_c[row * Hd + col];
          const float iv = sigm(ai[r] + bi);
          const float fv = sigm(af[r] + bf);
          const float gv = tanhf(ag[r] + bg);
          const float ov = sigm(ao[r] + bo);
          const float cn = fv * cold + iv * gv;
          const float hn = ov * tanhf(cn);
          sf_c[row * Hd + col] = cn;
          hnxt[row * Hd + col] = (_Float16)hn;
          mid[((size_t)t * B + b0 + row) * Hd + col] = (_Float16)hn;
        }
      }
    }
    __syncthreads();
    _Float16* tmp = hcur; hcur = hnxt; hnxt = tmp;
  }
}

// ---------------- mid LSTM, fused over Te ----------------
__global__ void __launch_bounds__(256)
mid_kernel(const _Float16* __restrict__ midin,   // [Te,B,H]
           const _Float16* __restrict__ Wih16,   // [4H,H]
           const _Float16* __restrict__ Whh16,   // [4H,H]
           const float* __restrict__ mb,         // [4H]
           _Float16* __restrict__ midout,        // [B,Te,H]
           float* __restrict__ hN, float* __restrict__ cN,  // [B,H]
           int B, int Te) {
  extern __shared__ __align__(16) char smem[];
  _Float16* sh_h0 = (_Float16*)smem;
  _Float16* sh_h1 = sh_h0 + BT * Hd;
  float* sf_c = (float*)(sh_h1 + BT * Hd);

  const int tid = threadIdx.x;
  const int w = tid >> 5;
  const int lane = tid & 31;
  const int nn = lane & 15;
  const int lg = lane >> 4;
  const int b0 = blockIdx.x * BT;

  for (int i = tid; i < BT * Hd; i += 256) { sh_h0[i] = (_Float16)0.0f; sf_c[i] = 0.0f; }
  __syncthreads();
  _Float16* hcur = sh_h0;
  _Float16* hnxt = sh_h1;

  for (int t = 0; t < Te; ++t) {
    const _Float16* xt = midin + ((size_t)t * B + b0) * Hd;   // rows contiguous
    const int u0 = w * 16;
    for (int rt = 0; rt < 8; ++rt) {
      v8f ai = {}, af = {}, ag = {}, ao = {};
#pragma unroll
      for (int k0 = 0; k0 < Hd; k0 += 32) {
        v16h a = frag_a_h(xt + rt * 16 * Hd + k0, Hd);        // A direct from L2
        ai = wmma16(a, frag_bT(Wih16 + (0 * Hd + u0) * Hd + k0, Hd), ai);
        af = wmma16(a, frag_bT(Wih16 + (1 * Hd + u0) * Hd + k0, Hd), af);
        ag = wmma16(a, frag_bT(Wih16 + (2 * Hd + u0) * Hd + k0, Hd), ag);
        ao = wmma16(a, frag_bT(Wih16 + (3 * Hd + u0) * Hd + k0, Hd), ao);
      }
#pragma unroll
      for (int k0 = 0; k0 < Hd; k0 += 32) {
        v16h a = frag_a_h(hcur + rt * 16 * Hd + k0, Hd);
        ai = wmma16(a, frag_bT(Whh16 + (0 * Hd + u0) * Hd + k0, Hd), ai);
        af = wmma16(a, frag_bT(Whh16 + (1 * Hd + u0) * Hd + k0, Hd), af);
        ag = wmma16(a, frag_bT(Whh16 + (2 * Hd + u0) * Hd + k0, Hd), ag);
        ao = wmma16(a, frag_bT(Whh16 + (3 * Hd + u0) * Hd + k0, Hd), ao);
      }
      const float bi = mb[0 * Hd + u0 + nn];
      const float bf = mb[1 * Hd + u0 + nn];
      const float bg = mb[2 * Hd + u0 + nn];
      const float bo = mb[3 * Hd + u0 + nn];
#pragma unroll
      for (int r = 0; r < 8; ++r) {
        const int row = rt * 16 + r + 8 * lg;
        const int col = u0 + nn;
        const float cold = sf_c[row * Hd + col];
        const float iv = sigm(ai[r] + bi);
        const float fv = sigm(af[r] + bf);
        const float gv = tanhf(ag[r] + bg);
        const float ov = sigm(ao[r] + bo);
        const float cn = fv * cold + iv * gv;
        const float hn = ov * tanhf(cn);
        sf_c[row * Hd + col] = cn;
        hnxt[row * Hd + col] = (_Float16)hn;
        midout[((size_t)(b0 + row) * Te + t) * Hd + col] = (_Float16)hn;
        if (t == Te - 1) {
          hN[(size_t)(b0 + row) * Hd + col] = hn;
          cN[(size_t)(b0 + row) * Hd + col] = cn;
        }
      }
    }
    __syncthreads();
    _Float16* tmp = hcur; hcur = hnxt; hnxt = tmp;
  }
}

// ---------------- wx_mid = mid_out @ Wx^T + Wx_b  (one-shot GEMM) ----------------
__global__ void __launch_bounds__(256)
wx_kernel(const _Float16* __restrict__ midout,   // [rows,H]
          const _Float16* __restrict__ Wx16,     // [H,H]
          const float* __restrict__ Wxb,         // [H]
          _Float16* __restrict__ wx, long rows) {
  const int tid = threadIdx.x;
  const int w = tid >> 5;
  const int lane = tid & 31;
  const int nn = lane & 15;
  const int lg = lane >> 4;
  const long r0 = (long)blockIdx.x * BT;
  const int n0 = w * 16;
  for (int rt = 0; rt < 8; ++rt) {
    const long row0 = r0 + rt * 16;
    if (row0 >= rows) break;
    v8f acc = {};
#pragma unroll
    for (int k0 = 0; k0 < Hd; k0 += 32)
      acc = wmma16(frag_a_h(midout + row0 * Hd + k0, Hd),
                   frag_bT(Wx16 + n0 * Hd + k0, Hd), acc);
    const float bias = Wxb[n0 + nn];
#pragma unroll
    for (int r = 0; r < 8; ++r) {
      const long row = row0 + r + 8 * lg;
      wx[row * Hd + n0 + nn] = (_Float16)(acc[r] + bias);
    }
  }
}

// ---------------- decoder: temporal-attention LSTMCell, fused over Td ----------------
__global__ void __launch_bounds__(256)
dec_kernel(const _Float16* __restrict__ midout,  // [B,Te,H]
           const _Float16* __restrict__ wx,      // [B,Te,H]
           const float* __restrict__ hN, const float* __restrict__ cN,
           const _Float16* __restrict__ Wh16,    // [H,2H]
           const float* __restrict__ Vw, const float* __restrict__ Vb,
           const _Float16* __restrict__ Wih16, const _Float16* __restrict__ Whh16,
           const float* __restrict__ db,
           const float* __restrict__ regw, const float* __restrict__ regb,
           float* __restrict__ out,              // [B,Td]
           int B, int Te, int Td) {
  extern __shared__ __align__(16) char smem[];
  _Float16* sh_h0 = (_Float16*)smem;
  _Float16* sh_h1 = sh_h0 + BT * Hd;
  float* sf_c = (float*)(sh_h1 + BT * Hd);
  float* sf_q = sf_c + BT * Hd;
  float* sf_d = sf_q + BT * Hd;

  const int tid = threadIdx.x;
  const int w = tid >> 5;
  const int lane = tid & 31;
  const int nn = lane & 15;
  const int lg = lane >> 4;
  const int b0 = blockIdx.x * BT;
  const int row2 = tid >> 1;   // 0..127 (two threads per batch row)
  const int half = tid & 1;
  const int jb = half * 64;

  for (int i = tid; i < BT * Hd; i += 256) {
    sh_h0[i] = (_Float16)hN[(size_t)b0 * Hd + i];
    sf_c[i] = cN[(size_t)b0 * Hd + i];
  }
  __syncthreads();
  _Float16* hcur = sh_h0;
  _Float16* hnxt = sh_h1;

  for (int td = 0; td < Td; ++td) {
    // phase 1: q = [h,c] @ Wh^T  (no bias)
    {
      const int n0 = w * 16;
      for (int rt = 0; rt < 8; ++rt) {
        v8f acc = {};
#pragma unroll
        for (int k0 = 0; k0 < Hd; k0 += 32)
          acc = wmma16(frag_a_h(hcur + rt * 16 * Hd + k0, Hd),
                       frag_bT(Wh16 + n0 * (2 * Hd) + k0, 2 * Hd), acc);
#pragma unroll
        for (int k0 = 0; k0 < Hd; k0 += 32)
          acc = wmma16(frag_a_f(sf_c + rt * 16 * Hd + k0, Hd),
                       frag_bT(Wh16 + n0 * (2 * Hd) + Hd + k0, 2 * Hd), acc);
#pragma unroll
        for (int r = 0; r < 8; ++r)
          sf_q[(rt * 16 + r + 8 * lg) * Hd + n0 + nn] = acc[r];
      }
    }
    __syncthreads();

    // phase 2: s_t = tanh(q + wx_t) . V + Vb ; dec_in += s_t * mid_t (streams L2)
    {
      for (int j = 0; j < 64; ++j) sf_d[row2 * Hd + jb + j] = 0.0f;
      const float vbias = Vb[0];
      for (int t = 0; t < Te; ++t) {
        const _Float16* wxp = wx + ((size_t)(b0 + row2) * Te + t) * Hd + jb;
        float part = 0.0f;
        for (int j = 0; j < 64; ++j)
          part += tanhf(sf_q[row2 * Hd + jb + j] + (float)wxp[j]) * Vw[jb + j];
        float s = part + __shfl_xor(part, 1, 32) + vbias;
        const _Float16* mp = midout + ((size_t)(b0 + row2) * Te + t) * Hd + jb;
        for (int j = 0; j < 64; ++j)
          sf_d[row2 * Hd + jb + j] += s * (float)mp[j];
      }
    }
    __syncthreads();

    // phase 3: LSTM cell on dec_in
    {
      const int u0 = w * 16;
      for (int rt = 0; rt < 8; ++rt) {
        v8f ai = {}, af = {}, ag = {}, ao = {};
#pragma unroll
        for (int k0 = 0; k0 < Hd; k0 += 32) {
          v16h a = frag_a_f(sf_d + rt * 16 * Hd + k0, Hd);
          ai = wmma16(a, frag_bT(Wih16 + (0 * Hd + u0) * Hd + k0, Hd), ai);
          af = wmma16(a, frag_bT(Wih16 + (1 * Hd + u0) * Hd + k0, Hd), af);
          ag = wmma16(a, frag_bT(Wih16 + (2 * Hd + u0) * Hd + k0, Hd), ag);
          ao = wmma16(a, frag_bT(Wih16 + (3 * Hd + u0) * Hd + k0, Hd), ao);
        }
#pragma unroll
        for (int k0 = 0; k0 < Hd; k0 += 32) {
          v16h a = frag_a_h(hcur + rt * 16 * Hd + k0, Hd);
          ai = wmma16(a, frag_bT(Whh16 + (0 * Hd + u0) * Hd + k0, Hd), ai);
          af = wmma16(a, frag_bT(Whh16 + (1 * Hd + u0) * Hd + k0, Hd), af);
          ag = wmma16(a, frag_bT(Whh16 + (2 * Hd + u0) * Hd + k0, Hd), ag);
          ao = wmma16(a, frag_bT(Whh16 + (3 * Hd + u0) * Hd + k0, Hd), ao);
        }
        const float bi = db[0 * Hd + u0 + nn];
        const float bf = db[1 * Hd + u0 + nn];
        const float bg = db[2 * Hd + u0 + nn];
        const float bo = db[3 * Hd + u0 + nn];
#pragma unroll
        for (int r = 0; r < 8; ++r) {
          const int row = rt * 16 + r + 8 * lg;
          const int col = u0 + nn;
          const float cold = sf_c[row * Hd + col];
          const float iv = sigm(ai[r] + bi);
          const float fv = sigm(af[r] + bf);
          const float gv = tanhf(ag[r] + bg);
          const float ov = sigm(ao[r] + bo);
          const float cn = fv * cold + iv * gv;
          const float hn = ov * tanhf(cn);
          sf_c[row * Hd + col] = cn;
          hnxt[row * Hd + col] = (_Float16)hn;
        }
      }
    }
    __syncthreads();

    // phase 4: out[b,td] = h . reg_w + reg_b
    {
      float part = 0.0f;
      for (int j = 0; j < 64; ++j)
        part += (float)hnxt[row2 * Hd + jb + j] * regw[jb + j];
      float tot = part + __shfl_xor(part, 1, 32);
      if (half == 0) out[(size_t)(b0 + row2) * Td + td] = tot + regb[0];
    }
    __syncthreads();
    _Float16* tmp = hcur; hcur = hnxt; hnxt = tmp;
  }
}

extern "C" void kernel_launch(void* const* d_in, const int* in_sizes, int n_in,
                              void* d_out, int out_size, void* d_ws, size_t ws_size,
                              hipStream_t stream) {
  (void)in_sizes; (void)n_in; (void)out_size; (void)ws_size;
  const int B = 1024, Te = 168, Td = 24;

  const float* input = (const float*)d_in[0];
  const float* Wi_w = (const float*)d_in[2];
  const float* Wi_b = (const float*)d_in[3];
  const float* We_w = (const float*)d_in[4];
  const float* Vd_w = (const float*)d_in[5];
  const float* Vd_b = (const float*)d_in[6];
  const float* eWih = (const float*)d_in[7];
  const float* eWhh = (const float*)d_in[8];
  const float* ebih = (const float*)d_in[9];
  const float* ebhh = (const float*)d_in[10];
  const float* mWih = (const float*)d_in[11];
  const float* mWhh = (const float*)d_in[12];
  const float* mbih = (const float*)d_in[13];
  const float* mbhh = (const float*)d_in[14];
  const float* Wx_w = (const float*)d_in[15];
  const float* Wx_b = (const float*)d_in[16];
  const float* Wh_w = (const float*)d_in[17];
  const float* V_w  = (const float*)d_in[18];
  const float* V_b  = (const float*)d_in[19];
  const float* dWih = (const float*)d_in[20];
  const float* dWhh = (const float*)d_in[21];
  const float* dbih = (const float*)d_in[22];
  const float* dbhh = (const float*)d_in[23];
  const float* regw = (const float*)d_in[24];
  const float* regb = (const float*)d_in[25];

  size_t off = 0;
  auto alloc = [&](size_t bytes) -> void* {
    void* p = (char*)d_ws + off;
    off = (off + bytes + 255) & ~(size_t)255;
    return p;
  };
  _Float16* Wi16   = (_Float16*)alloc((size_t)128 * 64 * 2);
  _Float16* We16   = (_Float16*)alloc((size_t)128 * 256 * 2);
  _Float16* Vd16   = (_Float16*)alloc((size_t)64 * 128 * 2);
  _Float16* eWih16 = (_Float16*)alloc((size_t)512 * 64 * 2);
  _Float16* eWhh16 = (_Float16*)alloc((size_t)512 * 128 * 2);
  _Float16* mWih16 = (_Float16*)alloc((size_t)512 * 128 * 2);
  _Float16* mWhh16 = (_Float16*)alloc((size_t)512 * 128 * 2);
  _Float16* Wx16   = (_Float16*)alloc((size_t)128 * 128 * 2);
  _Float16* Wh16   = (_Float16*)alloc((size_t)128 * 256 * 2);
  _Float16* dWih16 = (_Float16*)alloc((size_t)512 * 128 * 2);
  _Float16* dWhh16 = (_Float16*)alloc((size_t)512 * 128 * 2);
  float* eb = (float*)alloc(512 * 4);
  float* mb = (float*)alloc(512 * 4);
  float* db = (float*)alloc(512 * 4);
  float* hN = (float*)alloc((size_t)B * 128 * 4);
  float* cN = (float*)alloc((size_t)B * 128 * 4);
  _Float16* mid    = (_Float16*)alloc((size_t)Te * B * 128 * 2);
  _Float16* midout = (_Float16*)alloc((size_t)B * Te * 128 * 2);
  _Float16* wx     = (_Float16*)alloc((size_t)B * Te * 128 * 2);

  auto cvt = [&](const float* s, _Float16* d, int n) {
    cvt_f16_kern<<<(n + 255) / 256, 256, 0, stream>>>(s, d, n);
  };
  cvt(Wi_w, Wi16, 128 * 64);
  cvt(We_w, We16, 128 * 256);
  cvt(Vd_w, Vd16, 64 * 128);
  cvt(eWih, eWih16, 512 * 64);
  cvt(eWhh, eWhh16, 512 * 128);
  cvt(mWih, mWih16, 512 * 128);
  cvt(mWhh, mWhh16, 512 * 128);
  cvt(Wx_w, Wx16, 128 * 128);
  cvt(Wh_w, Wh16, 128 * 256);
  cvt(dWih, dWih16, 512 * 128);
  cvt(dWhh, dWhh16, 512 * 128);
  bias2_kern<<<2, 256, 0, stream>>>(ebih, ebhh, eb, 512);
  bias2_kern<<<2, 256, 0, stream>>>(mbih, mbhh, mb, 512);
  bias2_kern<<<2, 256, 0, stream>>>(dbih, dbhh, db, 512);

  const size_t smem_enc = (size_t)(2 * BT * Hd * 2 + BT * Hd * 4 + BT * Fd * 2 +
                                   BT * Hd * 2 + BT * Fd * 4 + BT * Fd * 2);   // 224 KB
  enc_kernel<<<B / BT, 256, smem_enc, stream>>>(input, Wi16, Wi_b, We16, Vd16, Vd_b,
                                                eWih16, eWhh16, eb, mid, B, Te);

  const size_t smem_mid = (size_t)(2 * BT * Hd * 2 + BT * Hd * 4);             // 128 KB
  mid_kernel<<<B / BT, 256, smem_mid, stream>>>(mid, mWih16, mWhh16, mb, midout,
                                                hN, cN, B, Te);

  wx_kernel<<<(B * Te) / BT, 256, 0, stream>>>(midout, Wx16, Wx_b, wx, (long)B * Te);

  const size_t smem_dec = (size_t)(2 * BT * Hd * 2 + 3 * BT * Hd * 4);         // 256 KB
  dec_kernel<<<B / BT, 256, smem_dec, stream>>>(midout, wx, hN, cN, Wh16, V_w, V_b,
                                                dWih16, dWhh16, db, regw, regb,
                                                (float*)d_out, B, Te, Td);
}
